// Attention_67611375173884
// MI455X (gfx1250) — matrix operand order repaired and checked by reading, not compiled, not run on hardware
//
#include <hip/hip_runtime.h>

// ---------------------------------------------------------------------------
// Gated multi-head attention (OpenFold Attention) for gfx1250 (MI455X, wave32)
// B=4, Q=K=1024, C=256, H=8, D=32
//  - all GEMMs via v_wmma_f32_16x16x32_bf16 (f32 accumulate)
//  - flash-attention streaming: 128MB pair_bias read exactly once
//  - V tiles staged to LDS via global_load_async_to_lds_b128 (ASYNCcnt),
//    double-buffered; all fragment loads ping-pong pipelined (no reg rotation)
//  - wave-local LDS transpose uses s_wait_dscnt instead of block barriers
// ---------------------------------------------------------------------------

#define BN   4
#define QN   1024
#define KN   1024
#define CN   256
#define HN   8
#define DN   32
#define HD   256   // H*D

typedef __attribute__((ext_vector_type(16))) __bf16        v16bf;
typedef __attribute__((ext_vector_type(8)))  float         v8f;
typedef __attribute__((ext_vector_type(8)))  unsigned int  v8u;

#if defined(__has_builtin) && __has_builtin(__builtin_amdgcn_s_wait_asynccnt)
#define WAIT_ASYNC(n) __builtin_amdgcn_s_wait_asynccnt(n)
#else
#define WAIT_ASYNC(n) asm volatile("s_wait_asynccnt %0" :: "i"(n) : "memory")
#endif

// Async copy of 16 bytes/lane from global to LDS (tracked by ASYNCcnt).
__device__ __forceinline__ void async_g2l_b128(unsigned lds_byte_off,
                                               unsigned long long gaddr) {
    asm volatile("global_load_async_to_lds_b128 %0, %1, off"
                 :: "v"(lds_byte_off), "v"(gaddr) : "memory");
}

// Wave-local LDS store->load ordering: DS ops of one wave issue in order for
// all 32 lanes, so cross-lane visibility only needs DScnt drained + a fence.
__device__ __forceinline__ void lds_fence_wave() {
    asm volatile("s_wait_dscnt 0x0" ::: "memory");
}

__device__ __forceinline__ unsigned short f2bf_bits(float f) {
    unsigned u = __builtin_bit_cast(unsigned, f);
    unsigned r = u + 0x7FFFu + ((u >> 16) & 1u);   // round-to-nearest-even
    return (unsigned short)(r >> 16);
}

__device__ __forceinline__ v8f wmma_bf16(v8u a, v8u b, v8f c) {
    return __builtin_amdgcn_wmma_f32_16x16x32_bf16(
        false, __builtin_bit_cast(v16bf, a),
        false, __builtin_bit_cast(v16bf, b),
        (short)0, c, false, false);
}

// 16-lane-group reductions (xor masks < 16 stay inside each half-wave group)
__device__ __forceinline__ float half_max(float v) {
    v = fmaxf(v, __shfl_xor(v, 1, 32));
    v = fmaxf(v, __shfl_xor(v, 2, 32));
    v = fmaxf(v, __shfl_xor(v, 4, 32));
    v = fmaxf(v, __shfl_xor(v, 8, 32));
    return v;
}
__device__ __forceinline__ float half_sum(float v) {
    v += __shfl_xor(v, 1, 32);
    v += __shfl_xor(v, 2, 32);
    v += __shfl_xor(v, 4, 32);
    v += __shfl_xor(v, 8, 32);
    return v;
}

// ---------------------------------------------------------------------------
// K0: f32 -> bf16 with optional scale (folds 1/sqrt(D) into wq)
// ---------------------------------------------------------------------------
__global__ void ofa_cvt_bf16(const float* __restrict__ src,
                             unsigned short* __restrict__ dst, int n,
                             float scale) {
    int i = blockIdx.x * blockDim.x + threadIdx.x;
    if (i < n) dst[i] = f2bf_bits(src[i] * scale);
}

// ---------------------------------------------------------------------------
// K1: fused projections.  blockIdx.z: 0=Q 1=K 2=V 3=Gate
//   out[row, col] = sum_c x[row,c] * w[col,c]
// One wave per 16x64 tile; fully unrolled ping-pong pipeline over K-steps.
// ---------------------------------------------------------------------------
__global__ void ofa_proj(const unsigned short* __restrict__ qx,
                         const unsigned short* __restrict__ kvx,
                         const unsigned short* __restrict__ wq,
                         const unsigned short* __restrict__ wk,
                         const unsigned short* __restrict__ wv,
                         const unsigned short* __restrict__ wg,
                         const float* __restrict__ bg,
                         unsigned short* __restrict__ qh,   // [B,H,Q,D] bf16
                         unsigned short* __restrict__ kh,   // [B,H,K,D] bf16
                         unsigned short* __restrict__ vh,   // [B,H,K,D] bf16
                         float* __restrict__ gsig)          // [B*Q, HD] f32
{
    const int lane = threadIdx.x & 31;
    const int mode = blockIdx.z;
    const unsigned short* x = (mode == 1 || mode == 2) ? kvx : qx;
    const unsigned short* w = (mode == 0) ? wq : (mode == 1) ? wk
                            : (mode == 2) ? wv : wg;

    const int m0 = blockIdx.y * 16;
    const int n0 = blockIdx.x * 64;
    const int mrow = m0 + (lane & 15);
    const int kb2  = (lane >> 4) * 4;        // unused (A-frag also uses nb2)
    const int nb2  = (lane >> 4) * 8;        // B-frag half offset (uints)
    const int ncol = lane & 15;

    const unsigned int* xrow = (const unsigned int*)(x + (size_t)mrow * CN);
    const unsigned int* wrow[4];
#pragma unroll
    for (int t = 0; t < 4; ++t)
        wrow[t] = (const unsigned int*)(w + (size_t)(n0 + t * 16 + ncol) * CN);

    v8f acc[4] = {v8f{}, v8f{}, v8f{}, v8f{}};
    v8u aF[2]; v8u bF[2][4];

    auto loadfrag = [&](int buf, int c0) {
        const int base = c0 >> 1;
#pragma unroll
        for (int j = 0; j < 8; ++j)
            aF[buf][j] = xrow[base + nb2 + j];
#pragma unroll
        for (int t = 0; t < 4; ++t)
#pragma unroll
            for (int j = 0; j < 8; ++j)
                bF[buf][t][j] = wrow[t][base + nb2 + j];
    };

    loadfrag(0, 0);
#pragma unroll
    for (int s = 0; s < CN / 32; ++s) {      // 8 steps, fully unrolled
        if (s + 1 < CN / 32) loadfrag((s + 1) & 1, (s + 1) * 32);
#pragma unroll
        for (int t = 0; t < 4; ++t)
            acc[t] = wmma_bf16(aF[s & 1], bF[s & 1][t], acc[t]);
    }

    const int rbase = (lane >> 4) * 8;
#pragma unroll
    for (int t = 0; t < 4; ++t) {
        const int col = n0 + t * 16 + ncol;
#pragma unroll
        for (int r = 0; r < 8; ++r) {
            const int row = m0 + rbase + r;            // [0, 4096)
            float v = acc[t][r];
            if (mode == 3) {
                float g = 1.0f / (1.0f + __expf(-(v + bg[col])));
                gsig[(size_t)row * HD + col] = g;
            } else {
                const int b = row >> 10, qi = row & 1023;
                const int h = col >> 5,  d  = col & 31;
                size_t o = (((size_t)(b * HN + h)) * QN + qi) * DN + d;
                unsigned short us = f2bf_bits(v);      // wq pre-scaled by 1/sqrt(D)
                if (mode == 0)      qh[o] = us;
                else if (mode == 1) kh[o] = us;
                else                vh[o] = us;
            }
        }
    }
}

// ---------------------------------------------------------------------------
// K2: flash attention.  Block = 128 thr (4 waves); one 16-query tile per wave.
// kk loop unrolled x2 so the ping-pong parity is a compile-time constant.
// ---------------------------------------------------------------------------
__global__ void ofa_attn(const unsigned short* __restrict__ qh,
                         const unsigned short* __restrict__ kh,
                         const unsigned short* __restrict__ vh,
                         const float* __restrict__ mask_bias,   // [B,K]
                         const float* __restrict__ pair_bias,   // [B,H,Q,K]
                         const float* __restrict__ gsig,        // [B*Q,HD]
                         unsigned short* __restrict__ og)       // [B*Q,HD] bf16
{
    __shared__ unsigned short Pb[4][16 * 32];       // per-wave P tile
    __shared__ unsigned short Vb[4][2][32 * DN];    // per-wave double-buffered V

    const int lane = threadIdx.x & 31;
    const int wid  = threadIdx.x >> 5;
    const int qt   = blockIdx.x * 4 + wid;
    const int h    = blockIdx.y;
    const int b    = blockIdx.z;

    const size_t head = ((size_t)(b * HN + h)) * QN;
    const int m     = lane & 15;
    const int kb2   = (lane >> 4) * 4;
    const int nb2   = (lane >> 4) * 8;
    const int ncol  = lane & 15;
    const int rbase = (lane >> 4) * 8;
    const int khalf = (lane >> 4) * 16;

    // Q A-fragment (registers for the whole kernel)
    const unsigned int* qrow =
        (const unsigned int*)(qh + (head + qt * 16 + m) * DN);
    v8u aq;
#pragma unroll
    for (int j = 0; j < 8; ++j)
        aq[j] = qrow[nb2 + j];

    v8f O0 = {}, O1 = {};
    float rm[8], rl[8];
#pragma unroll
    for (int r = 0; r < 8; ++r) { rm[r] = -1e30f; rl[r] = 0.0f; }

    const float* pbq = pair_bias + (head + qt * 16 + rbase) * (size_t)KN;

    v8u bkbuf[2][2];                 // ping-pong K-tile fragments
    auto load_bk = [&](int buf, int kk) {
#pragma unroll
        for (int t = 0; t < 2; ++t) {
            const unsigned int* krow = (const unsigned int*)
                (kh + (head + kk + t * 16 + ncol) * DN);
#pragma unroll
            for (int j = 0; j < 8; ++j) bkbuf[buf][t][j] = krow[nb2 + j];
        }
    };
    auto stage_v = [&](int buf, int kk) {   // async 2KB contiguous V tile
        unsigned lo = (unsigned)(uintptr_t)(&Vb[wid][buf][0]) + lane * 16;
        unsigned long long ga =
            (unsigned long long)(uintptr_t)(vh + (head + kk) * DN) + lane * 16;
#pragma unroll
        for (int i = 0; i < 4; ++i)
            async_g2l_b128(lo + i * 512, ga + i * 512);
    };

    // prime pipeline for kk = 0
    load_bk(0, 0);
    stage_v(0, 0);

    auto step = [&](int kk, int cur) {      // cur is a literal at call sites
        // ---- scores from preloaded fragments ----
        v8f s[2];
#pragma unroll
        for (int t = 0; t < 2; ++t) s[t] = wmma_bf16(aq, bkbuf[cur][t], v8f{});

        // ---- issue next step's K fragments + async V stage ----
        const int nkk = kk + 32;
        if (nkk < KN) {
            load_bk(cur ^ 1, nkk);
            stage_v(cur ^ 1, nkk);
            __builtin_prefetch(pbq + nkk + ncol, 0, 1);
        }

        // ---- biases (the 128MB pair_bias stream, read exactly once) ----
#pragma unroll
        for (int t = 0; t < 2; ++t) {
            const int key = kk + t * 16 + ncol;
            const float mb = mask_bias[b * KN + key];
            const float* pb = pbq + key;
#pragma unroll
            for (int r = 0; r < 8; ++r)
                s[t][r] += mb + pb[(size_t)r * KN];
        }

        // ---- online softmax (row = query, spread over 16-lane groups) ----
#pragma unroll
        for (int r = 0; r < 8; ++r) {
            float mx = half_max(fmaxf(s[0][r], s[1][r]));
            float nm = fmaxf(rm[r], mx);
            float sc = __expf(rm[r] - nm);
            rm[r] = nm;
            float p0 = __expf(s[0][r] - nm);
            float p1 = __expf(s[1][r] - nm);
            s[0][r] = p0; s[1][r] = p1;
            rl[r] = rl[r] * sc + half_sum(p0 + p1);
            O0[r] *= sc; O1[r] *= sc;
        }

        // ---- P: C-layout -> A-layout via wave-private LDS tile ----
        unsigned short* P = Pb[wid];
#pragma unroll
        for (int t = 0; t < 2; ++t)
#pragma unroll
            for (int r = 0; r < 8; ++r)
                P[(rbase + r) * 32 + t * 16 + ncol] = f2bf_bits(s[t][r]);
        lds_fence_wave();                   // same-wave DS order + DScnt drain
        const unsigned int* Pu = (const unsigned int*)P;
        v8u ap;
#pragma unroll
        for (int j = 0; j < 8; ++j)
            ap[j] = Pu[m * 16 + nb2 + j];

        // ---- wait for this step's async V tile (older 4 of <=8 in flight) --
        if (nkk < KN) WAIT_ASYNC(4);
        else          WAIT_ASYNC(0);

        // ---- PV: two 16x16 d-tiles, contraction over 32 keys (LDS gather) --
        const unsigned short* Vt = &Vb[wid][cur][0];
#pragma unroll
        for (int t = 0; t < 2; ++t) {
            const int d = t * 16 + ncol;
            v8u bv;
#pragma unroll
            for (int j = 0; j < 8; ++j) {
                unsigned lov = Vt[(khalf + 2 * j)     * DN + d];
                unsigned hiv = Vt[(khalf + 2 * j + 1) * DN + d];
                bv[j] = lov | (hiv << 16);
            }
            if (t == 0) O0 = wmma_bf16(ap, bv, O0);
            else        O1 = wmma_bf16(ap, bv, O1);
        }
    };

    for (int kk = 0; kk < KN; kk += 64) {   // x2 unroll: parity is constant
        step(kk, 0);
        step(kk + 32, 1);
    }

    // ---- normalize, gate, store bf16 ----
#pragma unroll
    for (int r = 0; r < 8; ++r) {
        float inv = 1.0f / rl[r];
        O0[r] *= inv; O1[r] *= inv;
    }
#pragma unroll
    for (int t = 0; t < 2; ++t)
#pragma unroll
        for (int r = 0; r < 8; ++r) {
            const int q   = qt * 16 + rbase + r;
            const int col = h * DN + t * 16 + ncol;
            size_t o = ((size_t)(b * QN + q)) * HD + col;
            float v = (t ? O1[r] : O0[r]) * gsig[o];
            og[o] = f2bf_bits(v);
        }
}

// ---------------------------------------------------------------------------
// K3: output projection  out[row,n] = sum_c og[row,c]*wo[n,c] + bo[n]  (f32)
// Same fully-unrolled ping-pong structure as K1.
// ---------------------------------------------------------------------------
__global__ void ofa_oproj(const unsigned short* __restrict__ og,
                          const unsigned short* __restrict__ wo,
                          const float* __restrict__ bo,
                          float* __restrict__ out)
{
    const int lane = threadIdx.x & 31;
    const int m0 = blockIdx.y * 16;
    const int n0 = blockIdx.x * 64;
    const int mrow = m0 + (lane & 15);
    const int kb2  = (lane >> 4) * 4;
    const int nb2  = (lane >> 4) * 8;
    const int ncol = lane & 15;

    const unsigned int* xrow = (const unsigned int*)(og + (size_t)mrow * HD);
    const unsigned int* wrow[4];
#pragma unroll
    for (int t = 0; t < 4; ++t)
        wrow[t] = (const unsigned int*)(wo + (size_t)(n0 + t * 16 + ncol) * HD);

    v8f acc[4] = {v8f{}, v8f{}, v8f{}, v8f{}};
    v8u aF[2]; v8u bF[2][4];

    auto loadfrag = [&](int buf, int c0) {
        const int base = c0 >> 1;
#pragma unroll
        for (int j = 0; j < 8; ++j)
            aF[buf][j] = xrow[base + nb2 + j];
#pragma unroll
        for (int t = 0; t < 4; ++t)
#pragma unroll
            for (int j = 0; j < 8; ++j)
                bF[buf][t][j] = wrow[t][base + nb2 + j];
    };

    loadfrag(0, 0);
#pragma unroll
    for (int s = 0; s < HD / 32; ++s) {
        if (s + 1 < HD / 32) loadfrag((s + 1) & 1, (s + 1) * 32);
#pragma unroll
        for (int t = 0; t < 4; ++t)
            acc[t] = wmma_bf16(aF[s & 1], bF[s & 1][t], acc[t]);
    }

    const int rbase = (lane >> 4) * 8;
#pragma unroll
    for (int t = 0; t < 4; ++t) {
        const int col = n0 + t * 16 + ncol;
#pragma unroll
        for (int r = 0; r < 8; ++r) {
            const int row = m0 + rbase + r;
            out[(size_t)row * CN + col] = acc[t][r] + bo[col];
        }
    }
}

// ---------------------------------------------------------------------------
// Host launcher
// ---------------------------------------------------------------------------
extern "C" void kernel_launch(void* const* d_in, const int* in_sizes, int n_in,
                              void* d_out, int out_size, void* d_ws, size_t ws_size,
                              hipStream_t stream) {
    const float* q_x   = (const float*)d_in[0];
    const float* kv_x  = (const float*)d_in[1];
    const float* maskb = (const float*)d_in[2];
    const float* pairb = (const float*)d_in[3];
    const float* wq    = (const float*)d_in[4];
    const float* wk    = (const float*)d_in[5];
    const float* wv    = (const float*)d_in[6];
    const float* wg    = (const float*)d_in[7];
    const float* bg    = (const float*)d_in[8];
    const float* wo    = (const float*)d_in[9];
    const float* bo    = (const float*)d_in[10];
    float* out = (float*)d_out;

    char* ws = (char*)d_ws;
    size_t off = 0;
    auto take = [&](size_t bytes) -> char* {
        char* p = ws + off;
        off = (off + bytes + 255) & ~(size_t)255;
        return p;
    };
    const size_t ACT = (size_t)BN * QN * CN;   // 1,048,576 elems
    const size_t WEL = (size_t)HD * CN;        // 65,536 elems

    unsigned short* qx_bf = (unsigned short*)take(ACT * 2);
    unsigned short* kv_bf = (unsigned short*)take(ACT * 2);
    unsigned short* wq_bf = (unsigned short*)take(WEL * 2);
    unsigned short* wk_bf = (unsigned short*)take(WEL * 2);
    unsigned short* wv_bf = (unsigned short*)take(WEL * 2);
    unsigned short* wg_bf = (unsigned short*)take(WEL * 2);
    unsigned short* wo_bf = (unsigned short*)take(WEL * 2);
    unsigned short* qh    = (unsigned short*)take((size_t)BN * HN * QN * DN * 2);
    unsigned short* kh    = (unsigned short*)take((size_t)BN * HN * KN * DN * 2);
    unsigned short* vh    = (unsigned short*)take((size_t)BN * HN * KN * DN * 2);
    float*          gsig  = (float*)take(ACT * 4);
    unsigned short* og    = (unsigned short*)take(ACT * 2);
    (void)ws_size; (void)n_in; (void)in_sizes; (void)out_size;

    const float qscale = 0.17677669529663687f;   // 1/sqrt(D), folded into wq

    ofa_cvt_bf16<<<(int)((ACT + 255) / 256), 256, 0, stream>>>(q_x,  qx_bf, (int)ACT, 1.0f);
    ofa_cvt_bf16<<<(int)((ACT + 255) / 256), 256, 0, stream>>>(kv_x, kv_bf, (int)ACT, 1.0f);
    ofa_cvt_bf16<<<(int)((WEL + 255) / 256), 256, 0, stream>>>(wq, wq_bf, (int)WEL, qscale);
    ofa_cvt_bf16<<<(int)((WEL + 255) / 256), 256, 0, stream>>>(wk, wk_bf, (int)WEL, 1.0f);
    ofa_cvt_bf16<<<(int)((WEL + 255) / 256), 256, 0, stream>>>(wv, wv_bf, (int)WEL, 1.0f);
    ofa_cvt_bf16<<<(int)((WEL + 255) / 256), 256, 0, stream>>>(wg, wg_bf, (int)WEL, 1.0f);
    ofa_cvt_bf16<<<(int)((WEL + 255) / 256), 256, 0, stream>>>(wo, wo_bf, (int)WEL, 1.0f);

    ofa_proj<<<dim3(4, 256, 4), 32, 0, stream>>>(
        qx_bf, kv_bf, wq_bf, wk_bf, wv_bf, wg_bf, bg, qh, kh, vh, gsig);

    ofa_attn<<<dim3(16, HN, BN), 128, 0, stream>>>(
        qh, kh, vh, maskb, pairb, gsig, og);

    ofa_oproj<<<dim3(4, 256), 32, 0, stream>>>(og, wo_bf, bo, out);
}
